// multi_head_attention_14955076125171
// MI455X (gfx1250) — matrix operand order repaired and checked by reading, not compiled, not run on hardware
//
#include <hip/hip_runtime.h>
#include <stdint.h>

// ---------------------------------------------------------------------------
// MI455X (gfx1250) multi-head attention, bf16 WMMA pipeline.
// D = A x B + C via V_WMMA_F32_16X16X32_BF16, wave32.
// ---------------------------------------------------------------------------

typedef __attribute__((ext_vector_type(16))) __bf16 v16bf;
typedef __attribute__((ext_vector_type(8)))  float  f32x8;

union ABFrag {
    v16bf v;
    uint4 q[2];
};

static __device__ __forceinline__ f32x8 wmma_bf16(const ABFrag& a, const ABFrag& b, f32x8 c) {
    return __builtin_amdgcn_wmma_f32_16x16x32_bf16(false, a.v, false, b.v, (short)0, c, false, false);
}

static __device__ __forceinline__ uint16_t bfbits(float x) {
    union { float f; uint32_t u; } v; v.f = x;
    uint32_t r = v.u + 0x7FFFu + ((v.u >> 16) & 1u);   // round-to-nearest-even
    return (uint16_t)(r >> 16);
}

static __device__ __forceinline__ f32x8 zero8() {
    f32x8 z;
#pragma unroll
    for (int i = 0; i < 8; ++i) z[i] = 0.0f;
    return z;
}

// ---------------------------------------------------------------------------
// fp32 -> bf16 cast, 4 elements/thread
// ---------------------------------------------------------------------------
__global__ __launch_bounds__(256) void cast_f32_to_bf16(const float* __restrict__ in,
                                                        uint16_t* __restrict__ out, int n4) {
    int i = blockIdx.x * 256 + threadIdx.x;
    if (i >= n4) return;
    float4 v = ((const float4*)in)[i];
    ushort4 o;
    o.x = bfbits(v.x); o.y = bfbits(v.y); o.z = bfbits(v.z); o.w = bfbits(v.w);
    ((ushort4*)out)[i] = o;
}

// ---------------------------------------------------------------------------
// C(M,1024) = A(M,1024) x W(1024,1024)^T + bias    (bf16 in, f32 accum)
// mode 0: write bf16, head-major [B,H,S,HD], value = (acc+bias)*scale
// mode 1: write f32 row-major (M,1024)
// Block: 256 thr = 8 waves; block tile 128x128; wave tile 64x32; K-step 32.
// ---------------------------------------------------------------------------
__global__ __launch_bounds__(256) void gemm_nt_bf16(
    const uint16_t* __restrict__ A,
    const uint16_t* __restrict__ W,
    const float*    __restrict__ bias,
    uint16_t*       __restrict__ outB,
    float*          __restrict__ outF,
    int mode, float scale)
{
    __shared__ uint16_t lA[128 * 40];   // padded stride: 40 halves = 80 B
    __shared__ uint16_t lW[128 * 40];

    const int tid  = threadIdx.x;
    const int lane = tid & 31;
    const int wv   = tid >> 5;
    const int wm   = (wv >> 2) * 64;          // wave M offset in block tile
    const int wn   = (wv & 3)  * 32;          // wave N offset in block tile
    const int r    = lane & 15;
    const int hiA  = (lane & 16) ? 8  : 0;    // A-frag K sub-base (halves)
    const int hiB  = (lane & 16) ? 16 : 0;    // unused; B-frags also load at hiA
    const int hi8  = (lane & 16) ? 8  : 0;    // C row offset

    const int mBase = blockIdx.x * 128;
    const int nBase = blockIdx.y * 128;

    f32x8 acc[4][2];
#pragma unroll
    for (int mt = 0; mt < 4; ++mt)
#pragma unroll
        for (int nt = 0; nt < 2; ++nt) acc[mt][nt] = zero8();

    for (int kt = 0; kt < 32; ++kt) {
        const int k0 = kt * 32;
#pragma unroll
        for (int i = 0; i < 2; ++i) {
            int c   = tid + i * 256;          // 0..511 chunks of 16B
            int row = c >> 2;
            int col = (c & 3) * 8;            // halves
            *(uint4*)(&lA[row * 40 + col]) =
                *(const uint4*)(&A[(size_t)(mBase + row) * 1024 + k0 + col]);
            *(uint4*)(&lW[row * 40 + col]) =
                *(const uint4*)(&W[(size_t)(nBase + row) * 1024 + k0 + col]);
        }
        __syncthreads();

        if (kt < 31) {                        // global_prefetch_b8 of next K tile
            int k1 = k0 + 32;
#pragma unroll
            for (int i = 0; i < 2; ++i) {
                int c   = tid + i * 256;
                int row = c >> 2;
                int col = (c & 3) * 8;
                __builtin_prefetch(&A[(size_t)(mBase + row) * 1024 + k1 + col]);
                __builtin_prefetch(&W[(size_t)(nBase + row) * 1024 + k1 + col]);
            }
        }

        ABFrag b[2];
#pragma unroll
        for (int nt = 0; nt < 2; ++nt) {
            const uint16_t* p = &lW[(wn + nt * 16 + r) * 40 + hiA];
            b[nt].q[0] = *(const uint4*)(p);
            b[nt].q[1] = *(const uint4*)(p + 16);
        }
#pragma unroll
        for (int mt = 0; mt < 4; ++mt) {
            ABFrag a;
            const uint16_t* p = &lA[(wm + mt * 16 + r) * 40 + hiA];
            a.q[0] = *(const uint4*)(p);
            a.q[1] = *(const uint4*)(p + 16);
#pragma unroll
            for (int nt = 0; nt < 2; ++nt)
                acc[mt][nt] = wmma_bf16(a, b[nt], acc[mt][nt]);
        }
        __syncthreads();
    }

#pragma unroll
    for (int mt = 0; mt < 4; ++mt) {
#pragma unroll
        for (int nt = 0; nt < 2; ++nt) {
            int n = nBase + wn + nt * 16 + r;
            float bv = bias[n];
#pragma unroll
            for (int i = 0; i < 8; ++i) {
                int m = mBase + wm + mt * 16 + hi8 + i;
                float v = (acc[mt][nt][i] + bv) * scale;
                if (mode == 0) {
                    int h = n >> 6, d = n & 63;
                    int bb = m >> 11, s = m & 2047;
                    outB[(((size_t)(bb * 16 + h)) * 2048 + s) * 64 + d] = bfbits(v);
                } else {
                    outF[(size_t)m * 1024 + n] = v;
                }
            }
        }
    }
}

// ---------------------------------------------------------------------------
// Flash attention per (b,h): Q,K,V are [B*H][2048][64] bf16 (Q pre-scaled
// by 1/sqrt(64)). Writes Ctx in the same head-major layout == reference's
// un-permuted reshape. 128 queries per block, 64-key tiles, online softmax.
// ---------------------------------------------------------------------------
__global__ __launch_bounds__(256) void flash_attn_bf16(
    const uint16_t* __restrict__ Qh,
    const uint16_t* __restrict__ Kh,
    const uint16_t* __restrict__ Vh,
    uint16_t*       __restrict__ Ctx)
{
    __shared__ uint16_t sQ[128 * 72];   // query tile   (row, hd), stride 72
    __shared__ uint16_t sK[64 * 72];    // key tile     (key, hd)
    __shared__ uint16_t sV[64 * 72];    // V transposed (hd, key)
    __shared__ uint16_t sP[128 * 72];   // probs bounce (row, key)

    const int tid  = threadIdx.x;
    const int lane = tid & 31;
    const int w    = tid >> 5;            // 8 waves x 16 query rows
    const int r    = lane & 15;
    const int hiA  = (lane & 16) ? 8  : 0;
    const int hiB  = (lane & 16) ? 16 : 0;
    const int hi8  = hiA;

    const int bh = blockIdx.y;            // b*16 + h
    const int q0 = blockIdx.x * 128;
    const uint16_t* qp    = Qh + ((size_t)bh * 2048 + q0) * 64;
    const uint16_t* kbase = Kh + (size_t)bh * 2048 * 64;
    const uint16_t* vbase = Vh + (size_t)bh * 2048 * 64;

#pragma unroll
    for (int i = 0; i < 4; ++i) {         // load 128x64 Q tile
        int c = tid + i * 256;
        int row = c >> 3;
        int col = (c & 7) * 8;
        *(uint4*)(&sQ[row * 72 + col]) = *(const uint4*)(&qp[(size_t)row * 64 + col]);
    }

    f32x8 o[4];
    float mrow[8], lrow[8];
#pragma unroll
    for (int nt = 0; nt < 4; ++nt) o[nt] = zero8();
#pragma unroll
    for (int i = 0; i < 8; ++i) { mrow[i] = -3.0e38f; lrow[i] = 0.0f; }

    for (int it = 0; it < 32; ++it) {
        const uint16_t* kp = kbase + (size_t)it * 64 * 64;
        const uint16_t* vp = vbase + (size_t)it * 64 * 64;
#pragma unroll
        for (int i = 0; i < 2; ++i) {     // 64x64 K tile + transposed V tile
            int c = tid + i * 256;
            int row = c >> 3;             // key index
            int col = (c & 7) * 8;        // hd base
            *(uint4*)(&sK[row * 72 + col]) = *(const uint4*)(&kp[(size_t)row * 64 + col]);
            uint4 vv = *(const uint4*)(&vp[(size_t)row * 64 + col]);
            const uint16_t* pv = (const uint16_t*)&vv;
#pragma unroll
            for (int j = 0; j < 8; ++j) sV[(col + j) * 72 + row] = pv[j];
        }
        __syncthreads();

        // ---- scores: S = Q(16x64) @ K^T (64 keys), 4 n-tiles x 2 K-steps
        f32x8 sc[4];
#pragma unroll
        for (int nt = 0; nt < 4; ++nt) sc[nt] = zero8();
#pragma unroll
        for (int kk = 0; kk < 64; kk += 32) {
            ABFrag aq;
            const uint16_t* p = &sQ[(w * 16 + r) * 72 + kk + hiA];
            aq.q[0] = *(const uint4*)(p);
            aq.q[1] = *(const uint4*)(p + 16);
#pragma unroll
            for (int nt = 0; nt < 4; ++nt) {
                ABFrag bk;
                const uint16_t* pb = &sK[(nt * 16 + r) * 72 + kk + hiA];
                bk.q[0] = *(const uint4*)(pb);
                bk.q[1] = *(const uint4*)(pb + 16);
                sc[nt] = wmma_bf16(aq, bk, sc[nt]);
            }
        }

        // ---- online softmax (row = hi8+i, cols spread over 16-lane half)
        float alpha[8];
#pragma unroll
        for (int i = 0; i < 8; ++i) {
            float t = sc[0][i];
            t = fmaxf(t, sc[1][i]); t = fmaxf(t, sc[2][i]); t = fmaxf(t, sc[3][i]);
#pragma unroll
            for (int off = 8; off >= 1; off >>= 1) t = fmaxf(t, __shfl_xor(t, off, 16));
            float mn = fmaxf(mrow[i], t);
            alpha[i] = __expf(mrow[i] - mn);
            mrow[i]  = mn;
        }
        float rsum[8];
#pragma unroll
        for (int i = 0; i < 8; ++i) rsum[i] = 0.0f;
#pragma unroll
        for (int nt = 0; nt < 4; ++nt)
#pragma unroll
            for (int i = 0; i < 8; ++i) {
                float e = __expf(sc[nt][i] - mrow[i]);
                sc[nt][i] = e;
                rsum[i] += e;
            }
#pragma unroll
        for (int i = 0; i < 8; ++i) {
            float t = rsum[i];
#pragma unroll
            for (int off = 8; off >= 1; off >>= 1) t += __shfl_xor(t, off, 16);
            lrow[i] = lrow[i] * alpha[i] + t;
        }
#pragma unroll
        for (int nt = 0; nt < 4; ++nt)
#pragma unroll
            for (int i = 0; i < 8; ++i) o[nt][i] *= alpha[i];

        // ---- bounce P through LDS: C-layout -> A-fragment layout (bf16)
#pragma unroll
        for (int nt = 0; nt < 4; ++nt)
#pragma unroll
            for (int i = 0; i < 8; ++i)
                sP[(w * 16 + hi8 + i) * 72 + nt * 16 + r] = bfbits(sc[nt][i]);

        // ---- O += P(16x64) @ V(64x64), V^T in LDS so B-frags contiguous
#pragma unroll
        for (int kk = 0; kk < 64; kk += 32) {
            ABFrag ap;
            const uint16_t* p = &sP[(w * 16 + r) * 72 + kk + hiA];
            ap.q[0] = *(const uint4*)(p);
            ap.q[1] = *(const uint4*)(p + 16);
#pragma unroll
            for (int nt = 0; nt < 4; ++nt) {
                ABFrag bv;
                const uint16_t* pb = &sV[(nt * 16 + r) * 72 + kk + hiA];
                bv.q[0] = *(const uint4*)(pb);
                bv.q[1] = *(const uint4*)(pb + 16);
                o[nt] = wmma_bf16(ap, bv, o[nt]);
            }
        }
        __syncthreads();
    }

    // ---- normalize, write ctx head-major (matches reference's reshape)
#pragma unroll
    for (int nt = 0; nt < 4; ++nt)
#pragma unroll
        for (int i = 0; i < 8; ++i) {
            float v = o[nt][i] / lrow[i];
            int row = q0 + w * 16 + hi8 + i;
            int dc  = nt * 16 + r;
            Ctx[((size_t)bh * 2048 + row) * 64 + dc] = bfbits(v);
        }
}

// ---------------------------------------------------------------------------
// Launcher. Workspace: Xb(8M) Wq/Wk/Wv/Wo(2M ea) Q(8M) K(8M) V(8M) Ctx(8M)
// ---------------------------------------------------------------------------
extern "C" void kernel_launch(void* const* d_in, const int* in_sizes, int n_in,
                              void* d_out, int out_size, void* d_ws, size_t ws_size,
                              hipStream_t stream) {
    (void)in_sizes; (void)n_in; (void)out_size; (void)ws_size;

    const float* X  = (const float*)d_in[0];
    const float* Wq = (const float*)d_in[1];
    const float* bq = (const float*)d_in[2];
    const float* Wk = (const float*)d_in[3];
    const float* bk = (const float*)d_in[4];
    const float* Wv = (const float*)d_in[5];
    const float* bv = (const float*)d_in[6];
    const float* Wo = (const float*)d_in[7];
    const float* bo = (const float*)d_in[8];

    char* ws = (char*)d_ws;
    uint16_t* Xb  = (uint16_t*)(ws);
    uint16_t* Wqb = (uint16_t*)(ws + ( 8ull << 20));
    uint16_t* Wkb = (uint16_t*)(ws + (10ull << 20));
    uint16_t* Wvb = (uint16_t*)(ws + (12ull << 20));
    uint16_t* Wob = (uint16_t*)(ws + (14ull << 20));
    uint16_t* Qb  = (uint16_t*)(ws + (16ull << 20));
    uint16_t* Kb  = (uint16_t*)(ws + (24ull << 20));
    uint16_t* Vb  = (uint16_t*)(ws + (32ull << 20));
    uint16_t* Cx  = (uint16_t*)(ws + (40ull << 20));

    cast_f32_to_bf16<<<4096, 256, 0, stream>>>(X,  Xb,  1048576);  // 4096x1024
    cast_f32_to_bf16<<<1024, 256, 0, stream>>>(Wq, Wqb, 262144);   // 1024x1024
    cast_f32_to_bf16<<<1024, 256, 0, stream>>>(Wk, Wkb, 262144);
    cast_f32_to_bf16<<<1024, 256, 0, stream>>>(Wv, Wvb, 262144);
    cast_f32_to_bf16<<<1024, 256, 0, stream>>>(Wo, Wob, 262144);

    dim3 g(32, 8);   // M/128 x N/128
    gemm_nt_bf16<<<g, 256, 0, stream>>>(Xb, Wqb, bq, Qb, nullptr, 0, 0.125f); // Q * 1/sqrt(64)
    gemm_nt_bf16<<<g, 256, 0, stream>>>(Xb, Wkb, bk, Kb, nullptr, 0, 1.0f);
    gemm_nt_bf16<<<g, 256, 0, stream>>>(Xb, Wvb, bv, Vb, nullptr, 0, 1.0f);

    flash_attn_bf16<<<dim3(16, 32), 256, 0, stream>>>(Qb, Kb, Vb, Cx);

    gemm_nt_bf16<<<g, 256, 0, stream>>>(Cx, Wob, bo, nullptr, (float*)d_out, 1, 1.0f);
}
